// VGG16Approx_66030827209201
// MI455X (gfx1250) — compile-verified, matrix operand order checked by reading, not run on hardware
//
#include <hip/hip_runtime.h>
#include <hip/hip_bf16.h>
#include <stdint.h>

// ---------------------------------------------------------------------------
// Types
// ---------------------------------------------------------------------------
typedef _Float16 f16;
typedef f16          v16h __attribute__((ext_vector_type(16)));
typedef float        v8f  __attribute__((ext_vector_type(8)));
typedef unsigned int u32;
typedef u32          v4u  __attribute__((ext_vector_type(4)));
typedef int          v8i  __attribute__((ext_vector_type(8)));
typedef int          v4i  __attribute__((ext_vector_type(4)));

#define CO_TILE 16      // output channels per block
#define CK_MAX  64      // input-channel chunk staged in LDS (16*64*9*4B = 36 KB)

// ---------------------------------------------------------------------------
// TDM: 2-D tile load (global -> LDS). D# packing per CDNA5 ISA ch.8.
//   tile: tile_d1 rows x tile_d0 elements (f32), row stride = stride_elems.
// Issued by one wave; EXEC is ignored by TENSOR_LOAD_TO_LDS.
// This toolchain declares the 6-arg builtin:
//   (uint32x4 g0, int32x8 g1, int32x4 g2, int32x4 g3, int32x8 g4, i32 cpol)
// ---------------------------------------------------------------------------
__device__ __forceinline__ void tdm_load_2d_f32(const void* gaddr, u32 lds_byte_off,
                                                u32 tile_d0, u32 tile_d1,
                                                u32 tensor_d0, u32 tensor_d1,
                                                u32 stride_elems)
{
#if __has_builtin(__builtin_amdgcn_tensor_load_to_lds)
    unsigned long long ga = (unsigned long long)(uintptr_t)gaddr;
    v4u g0;
    g0.x = 1u;                                        // count=1 (valid), user mode
    g0.y = lds_byte_off;                              // lds_addr [63:32]
    g0.z = (u32)(ga & 0xffffffffu);                   // global_addr[31:0]
    g0.w = (u32)((ga >> 32) & 0x01ffffffu)            // global_addr[56:32]
         | (2u << 30);                                // type=2 ("image")
    v8i g1;
    g1.s0 = (int)(2u << 16);                          // wg_mask=0, data_size=2 (4B)
    g1.s1 = (int)((tensor_d0 & 0xffffu) << 16);       // tensor_dim0[15:0] at bits 63:48
    g1.s2 = (int)((tensor_d0 >> 16) |                 // tensor_dim0[31:16]
                  ((tensor_d1 & 0xffffu) << 16));     // tensor_dim1[15:0]
    g1.s3 = (int)((tensor_d1 >> 16) |                 // tensor_dim1[31:16]
                  ((tile_d0 & 0xffffu) << 16));       // tile_dim0
    g1.s4 = (int)(tile_d1 & 0xffffu);                 // tile_dim1 (tile_dim2=0)
    g1.s5 = (int)stride_elems;                        // tensor_dim0_stride[31:0]
    g1.s6 = 0;                                        // stride[47:32], dim1_stride lo
    g1.s7 = 0;
    v4i z4 = {0, 0, 0, 0};                            // groups 2/3: dims unused
    v8i z8 = {0, 0, 0, 0, 0, 0, 0, 0};                // extra group (6-arg form)
    __builtin_amdgcn_tensor_load_to_lds(g0, g1, z4, z4, z8, 0);
#endif
}

// ---------------------------------------------------------------------------
// mean(|w|): deterministic two-pass reduction (no float atomics)
// ---------------------------------------------------------------------------
__global__ __launch_bounds__(256) void absmean_part_kernel(const float* __restrict__ w,
                                                           int n, float* __restrict__ part)
{
    __shared__ float sm[256];
    float s = 0.f;
    for (int i = blockIdx.x * 256 + threadIdx.x; i < n; i += 256 * gridDim.x)
        s += fabsf(w[i]);
    sm[threadIdx.x] = s;
    __syncthreads();
    #pragma unroll
    for (int off = 128; off > 0; off >>= 1) {
        if ((int)threadIdx.x < off) sm[threadIdx.x] += sm[threadIdx.x + off];
        __syncthreads();
    }
    if (threadIdx.x == 0) part[blockIdx.x] = sm[0];
}

__global__ __launch_bounds__(256) void absmean_final_kernel(const float* __restrict__ part,
                                                            float* __restrict__ mu_out)
{
    __shared__ float sm[256];
    sm[threadIdx.x] = part[threadIdx.x];
    __syncthreads();
    #pragma unroll
    for (int off = 128; off > 0; off >>= 1) {
        if ((int)threadIdx.x < off) sm[threadIdx.x] += sm[threadIdx.x + off];
        __syncthreads();
    }
    if (threadIdx.x == 0) mu_out[0] = sm[0];
}

// ---------------------------------------------------------------------------
// Min-plus pseudo-conv (+ReLU).  y[b,co,l] = relu(mu * sum_{ci,t} min(x_t, w_t))
// Block: 256 thr = 16 co-lanes x 16 pixel-lanes.  Weights staged in LDS by TDM.
// Grid: (Cout/16, ceil(L/16), B)
// ---------------------------------------------------------------------------
__global__ __launch_bounds__(256) void approx_conv_kernel(
    const float* __restrict__ x, const float* __restrict__ w,
    const float* __restrict__ mu_sum, float* __restrict__ y,
    int Cin, int Cout, int H, int W, float inv_numel)
{
    __shared__ float wsm[CO_TILE * CK_MAX * 9];   // 36 KB

    const int b       = blockIdx.z;
    const int co_base = blockIdx.x * CO_TILE;
    const int L       = H * W;
    const int tid     = threadIdx.x;
    const int co_idx  = tid >> 4;
    const int px_idx  = tid & 15;
    const int co      = co_base + co_idx;
    const int l       = blockIdx.y * 16 + px_idx;
    const int lc      = (l < L) ? l : 0;
    const float mu    = mu_sum[0] * inv_numel;

    const int oy = lc / W, ox = lc % W;
    const float* __restrict__ xb = x + (size_t)b * Cin * L;

    // Hoist the 3x3 tap offsets + validity (uniform over all channels).
    int  toff[9];
    bool tok[9];
    #pragma unroll
    for (int dy = 0; dy < 3; ++dy) {
        #pragma unroll
        for (int dx = 0; dx < 3; ++dx) {
            const int yy = oy + dy - 1, xx = ox + dx - 1;
            const bool ok = (yy >= 0) & (yy < H) & (xx >= 0) & (xx < W);
            tok[dy * 3 + dx]  = ok;
            toff[dy * 3 + dx] = ok ? (yy * W + xx) : 0;
        }
    }

    float acc = 0.f;
    const int CK = (Cin < CK_MAX) ? Cin : CK_MAX;

    for (int ck0 = 0; ck0 < Cin; ck0 += CK) {
#if __has_builtin(__builtin_amdgcn_tensor_load_to_lds)
        if (tid < 32) {   // one wave issues the DMA (TENSORcnt is per-wave)
            const float* gsrc = w + ((size_t)co_base * Cin + ck0) * 9;
            tdm_load_2d_f32(gsrc, (u32)(uintptr_t)&wsm[0],
                            (u32)(CK * 9), CO_TILE,
                            (u32)(Cin * 9), (u32)Cout, (u32)(Cin * 9));
            __builtin_amdgcn_s_wait_tensorcnt(0);
        }
#else
        for (int idx = tid; idx < CO_TILE * CK * 9; idx += 256) {
            const int r = idx / (CK * 9), c = idx % (CK * 9);
            wsm[idx] = w[((size_t)(co_base + r) * Cin + ck0) * 9 + c];
        }
#endif
        __syncthreads();

        const float* __restrict__ xc = xb + (size_t)ck0 * L;
        __builtin_prefetch(xc + (size_t)CK * L, 0, 1);   // global_prefetch_b8
        const float* __restrict__ wr = &wsm[co_idx * CK * 9];

        for (int ci = 0; ci < CK; ++ci) {
            const float* __restrict__ xci = xc + (size_t)ci * L;
            const float* __restrict__ wk  = wr + ci * 9;
            #pragma unroll
            for (int t = 0; t < 9; ++t) {
                const float v = tok[t] ? xci[toff[t]] : 0.f;
                acc += fminf(v, wk[t]);
            }
        }
        __syncthreads();   // protect LDS before next chunk overwrites it
    }

    if (l < L && co < Cout) {
        const float r = acc * mu;
        y[((size_t)b * Cout + co) * L + l] = (r > 0.f) ? r : 0.f;
    }
}

// ---------------------------------------------------------------------------
// 2x2 max-pool, stride 2
// ---------------------------------------------------------------------------
__global__ __launch_bounds__(256) void maxpool2_kernel(const float* __restrict__ x,
                                                       float* __restrict__ y,
                                                       int C, int H, int W)
{
    const int Ho = H >> 1, Wo = W >> 1;
    const size_t n = (size_t)32 * C * Ho * Wo;
    for (size_t i = (size_t)blockIdx.x * blockDim.x + threadIdx.x; i < n;
         i += (size_t)gridDim.x * blockDim.x) {
        const int wo = (int)(i % Wo);
        size_t t = i / Wo;
        const int ho = (int)(t % Ho); t /= Ho;
        const int c  = (int)(t % C);
        const int b  = (int)(t / C);
        const float* p = x + (((size_t)b * C + c) * H + (size_t)ho * 2) * W + (size_t)wo * 2;
        y[i] = fmaxf(fmaxf(p[0], p[1]), fmaxf(p[W], p[W + 1]));
    }
}

// ---------------------------------------------------------------------------
// FC layer with WMMA:  Y(32,N) = [relu](X(32,K) @ W(N,K)^T + bias)
// One wave computes one 16x16 tile.  K-loop unrolled by 2 with two
// independent WMMA accumulator chains (c0/c1) for matrix-pipe ILP.
// Requires K % 64 == 0 (holds: 512, 4096, 4096).
// ---------------------------------------------------------------------------
__device__ __forceinline__ void load_frag_f16(const float* __restrict__ row,
                                              int k0, int koff, v16h* frag)
{
    #pragma unroll
    for (int j = 0; j < 8; ++j) {
        (*frag)[j]     = (f16)row[k0 + koff + j];
        (*frag)[8 + j] = (f16)row[k0 + 16 + koff + j];
    }
}

__global__ __launch_bounds__(256) void fc_wmma_kernel(
    const float* __restrict__ X, const float* __restrict__ Wt,
    const float* __restrict__ bias, float* __restrict__ Y,
    int K, int N, int relu)
{
    const int wave = (int)((blockIdx.x * blockDim.x + threadIdx.x) >> 5);
    const int lane = (int)(threadIdx.x & 31);
    const int mtiles = 2;                       // M = 32
    const int ntiles = (N + 15) >> 4;
    if (wave >= mtiles * ntiles) return;        // wave-uniform: EXEC stays all-1s
    const int mt = wave % mtiles;
    const int nt = wave / mtiles;

    // 16-bit A 16x32 striping: lanes 0-15 hold K {0..7,16..23}, lanes 16-31
    // hold K {8..15,24..31} of row M = lane&15. B mirrors it over (N,K) rows.
    const int r    = lane & 15;
    const int koff = (lane >> 4) << 3;          // 0 or 8
    const int m    = mt * 16 + r;
    const int n    = nt * 16 + r;
    const float* __restrict__ xrow = X  + (size_t)m * K;
    const float* __restrict__ wrow = Wt + (size_t)((n < N) ? n : 0) * K;

    v8f c0 = {0.f, 0.f, 0.f, 0.f, 0.f, 0.f, 0.f, 0.f};
    v8f c1 = {0.f, 0.f, 0.f, 0.f, 0.f, 0.f, 0.f, 0.f};
    for (int k0 = 0; k0 < K; k0 += 64) {
        v16h a0, b0, a1, b1;
        load_frag_f16(xrow, k0,      koff, &a0);
        load_frag_f16(wrow, k0,      koff, &b0);
        load_frag_f16(xrow, k0 + 32, koff, &a1);
        load_frag_f16(wrow, k0 + 32, koff, &b1);
        c0 = __builtin_amdgcn_wmma_f32_16x16x32_f16(false, a0, false, b0,
                                                    (short)0, c0, false, false);
        c1 = __builtin_amdgcn_wmma_f32_16x16x32_f16(false, a1, false, b1,
                                                    (short)0, c1, false, false);
    }

    // D layout: VGPR v, lanes 0-15 -> M=v, N=lane; lanes 16-31 -> M=v+8, N=lane-16
    const int n_out = nt * 16 + (lane & 15);
    const int m_hi  = (lane >> 4) * 8;
    if (n_out < N) {
        const float bv = bias[n_out];
        #pragma unroll
        for (int v = 0; v < 8; ++v) {
            const int mm = mt * 16 + m_hi + v;
            float val = c0[v] + c1[v] + bv;
            if (relu) val = fmaxf(val, 0.f);
            Y[(size_t)mm * N + n_out] = val;
        }
    }
}

// ---------------------------------------------------------------------------
// Host-side orchestration
// ---------------------------------------------------------------------------
static const int  CH[13][2] = {{3,64},{64,64},{64,128},{128,128},{128,256},{256,256},
                               {256,256},{256,512},{512,512},{512,512},{512,512},
                               {512,512},{512,512}};
static const bool POOL[13]  = {false,true,false,true,false,false,true,
                               false,false,true,false,false,true};

extern "C" void kernel_launch(void* const* d_in, const int* in_sizes, int n_in,
                              void* d_out, int out_size, void* d_ws, size_t ws_size,
                              hipStream_t stream)
{
    (void)in_sizes; (void)n_in; (void)out_size; (void)ws_size;

    // Workspace layout (floats)
    float* bufA = (float*)d_ws;                 // 2,097,152 f (8 MB)
    float* bufB = bufA + 2097152;               // 8 MB
    float* part = bufB + 2097152;               // 256 f
    float* mu   = part + 256;                   // 16 f
    float* fcb1 = mu + 16;                      // 32*4096 f
    float* fcb2 = fcb1 + 32 * 4096;             // 32*4096 f

    const float* cur = (const float*)d_in[0];   // (32,3,32,32)
    int H = 32, W = 32;

    for (int i = 0; i < 13; ++i) {
        const int Cin = CH[i][0], Cout = CH[i][1];
        const float* w = (const float*)d_in[1 + i];
        const int numel = Cout * Cin * 9;

        absmean_part_kernel<<<256, 256, 0, stream>>>(w, numel, part);
        absmean_final_kernel<<<1, 256, 0, stream>>>(part, mu + i);

        const int L = H * W;
        float* nxt = (cur == bufA) ? bufB : bufA;
        dim3 grid((unsigned)(Cout / CO_TILE), (unsigned)((L + 15) / 16), 32u);
        approx_conv_kernel<<<grid, 256, 0, stream>>>(cur, w, mu + i, nxt,
                                                     Cin, Cout, H, W, 1.0f / (float)numel);
        cur = nxt;

        if (POOL[i]) {
            float* pdst = (cur == bufA) ? bufB : bufA;
            const size_t n = (size_t)32 * Cout * (H / 2) * (W / 2);
            int blocks = (int)((n + 255) / 256);
            if (blocks > 4096) blocks = 4096;
            maxpool2_kernel<<<blocks, 256, 0, stream>>>(cur, pdst, Cout, H, W);
            H >>= 1; W >>= 1;
            cur = pdst;
        }
    }

    // cur now holds (32,512,1,1) == X (32,512)
    auto launch_fc = [&](const float* Xp, const float* Wp, const float* Bp,
                         float* Yp, int K, int N, int relu) {
        const int waves  = 2 * ((N + 15) / 16);
        const int blocks = (waves * 32 + 255) / 256;
        fc_wmma_kernel<<<blocks, 256, 0, stream>>>(Xp, Wp, Bp, Yp, K, N, relu);
    };
    launch_fc(cur,  (const float*)d_in[14], (const float*)d_in[15], fcb1, 512,  4096, 1);
    launch_fc(fcb1, (const float*)d_in[16], (const float*)d_in[17], fcb2, 4096, 4096, 1);
    launch_fc(fcb2, (const float*)d_in[18], (const float*)d_in[19], (float*)d_out,
              4096, 10, 0);
}
